// MoLeRDecoder_1657857377022
// MI455X (gfx1250) — matrix operand order repaired and checked by reading, not compiled, not run on hardware
//
#include <hip/hip_runtime.h>
#include <hip/hip_bf16.h>
#include <math.h>

// ---------------------------------------------------------------------------
// MoLeR decoder forward pass for MI455X (gfx1250), wave32 + WMMA bf16.
//
// Input index map (setup_inputs dict order, then params flattened as a JAX
// pytree: dict keys sorted alphabetically, lists/tuples in order):
//   0 node_features (N,32) f32        1..3 adj_0..2 (E,2) i32
//   4 node_to_graph_map (N) i32       5 graph_to_focus_node_map (G) i32
//   6 candidate_attachment_points (A) 7 input_molecule_representations (G,512)
//   8 graphs_requiring_node_choices   9 candidate_edges (CE,2)
//   10 candidate_edge_features (CE,3) 11 num_graphs (scalar)
//   12..19  attach    [W(864,128) b W(128,64) b W(64,32) b W(32,1) b]
//   20..25  avg_score [W(96,128) b W(128,128) b W(128,8) b]
//   26..31  avg_trans [W(96,128) b W(128,128) b W(128,128) b]
//   32..37  dense     3x[W(64,64) b]
//   38..45  edge_scorer [W(963,128) b W(128,64) b W(64,32) b W(32,1) b]
//   46..53  edge_type   [W(963,128) b W(128,64) b W(64,32) b W(32,3) b]
//   54..65  ln_bias x12   66..77 ln_scale x12
//   78..113 msg[l][t] = 78+l*3+t  (W(64,64), no bias)
//   114     no_more_edges (99)
//   115..120 node_sel [W(768,256) b W(256,256) b W(256,65) b]
//   121..126 sum_score   127..132 sum_trans
//   133..134 w_init [W(32,64) b(64)]
// Outputs (flat): node_type_logits 2500*65 | edge_candidate_logits CE+G |
//                 edge_type_logits CE*3 | attachment logits A
// ---------------------------------------------------------------------------

#define NN 100000
#define FF 32
#define HH 64
#define LL 12
#define TT 3
#define EE 200000
#define GG 5000
#define MREPW 512
#define CEN 50000
#define APN 20000
#define NTL 65          // NT+1
#define NSEL 2500       // G//2
#define HEADS 8
#define HALFW 128
#define UBF 5.0f
#define EPSF 1e-7f

typedef __bf16 bf16_t;
typedef __attribute__((ext_vector_type(16))) __bf16 v16bf;
typedef __attribute__((ext_vector_type(8)))  float  v8f;

__device__ __forceinline__ v8f wmma_bf16(v16bf a, v16bf b, v8f c) {
  return __builtin_amdgcn_wmma_f32_16x16x32_bf16(false, a, false, b, (short)0, c, false, false);
}
// A-frag (16x32 bf16): lane L holds row M=L&15; slot i -> K index (ISA 7.12.2)
__device__ __forceinline__ int a_k(int i, int g) {
  int v = i >> 1, p = i & 1;
  return 16 * (v >> 2) + 2 * (v & 3) + p + 8 * g;
}
// B-frag (32x16 bf16): lane L holds col N=L&15; slot i -> same K as a_k
__device__ __forceinline__ int b_k(int i, int g) { return (i & 7) + 8 * g + 16 * (i >> 3); }

__device__ __forceinline__ float softplusf(float x) {
  return fmaxf(x, 0.f) + log1pf(expf(-fabsf(x)));
}
__device__ __forceinline__ void atomicMaxF(float* addr, float val) {
  unsigned* ua = (unsigned*)addr;
  unsigned old = *ua;
  while (true) {
    float f = __uint_as_float(old);
    if (f >= val) break;
    unsigned prev = atomicCAS(ua, old, __float_as_uint(val));
    if (prev == old) break;
    old = prev;
  }
}
// Speculative prefetch of a weight matrix into cache (global_prefetch_b8)
__device__ __forceinline__ void prefetchW(const float* p, int bytes) {
  for (int o = 0; o < bytes; o += 4096) __builtin_prefetch((const char*)p + o, 0, 1);
}

// Fully fused GEMM layer: K-outer / N-inner, A fragment loaded once per K-step,
// all C tiles held in VGPRs. A via per-lane fetch functor, C init functor,
// out -> LDS (full 16-wide tiles). act: 0 none, 1 relu, 2 leaky(0.2)
template <int KIN, int NOUT, class AF, class CF>
__device__ __forceinline__ void layer_fused(AF&& af, CF&& ci,
                                            const float* __restrict__ W,
                                            float* out, int act, int lane) {
  const int g = lane >> 4, n = lane & 15;
  constexpr int NTILES = NOUT / 16;
  v8f c[NTILES];
#pragma unroll
  for (int t = 0; t < NTILES; t++)
#pragma unroll
    for (int r = 0; r < 8; r++) c[t][r] = ci(r + 8 * g, t * 16 + n);
#pragma unroll
  for (int K0 = 0; K0 < KIN; K0 += 32) {
    v16bf a;
#pragma unroll
    for (int i = 0; i < 16; i++) a[i] = (bf16_t)af(K0 + a_k(i, g));
#pragma unroll
    for (int t = 0; t < NTILES; t++) {
      v16bf b;
#pragma unroll
      for (int i = 0; i < 16; i++)
        b[i] = (bf16_t)W[(size_t)(K0 + b_k(i, g)) * NOUT + t * 16 + n];
      c[t] = wmma_bf16(a, b, c[t]);
    }
  }
#pragma unroll
  for (int t = 0; t < NTILES; t++)
#pragma unroll
    for (int r = 0; r < 8; r++) {
      float x = c[t][r];
      if (act == 1) x = fmaxf(x, 0.f);
      else if (act == 2) x = (x > 0.f) ? x : 0.2f * x;
      out[(size_t)(r + 8 * g) * NOUT + t * 16 + n] = x;
    }
}

// Final GEMM layer -> global, runtime Kin/Nout, partial N tiles + row guard
template <class AF, class CF>
__device__ __forceinline__ void layer_out(AF&& af, CF&& ci, int Kin,
                                          const float* __restrict__ W, int Nout,
                                          float* out, int ld, int M0, int nrows,
                                          int act, int lane) {
  const int g = lane >> 4, n = lane & 15;
  for (int N0 = 0; N0 < Nout; N0 += 16) {
    int nv = (Nout - N0 < 16) ? (Nout - N0) : 16;
    v8f c;
#pragma unroll
    for (int r = 0; r < 8; r++) c[r] = (n < nv) ? ci(r + 8 * g, N0 + n) : 0.f;
    for (int K0 = 0; K0 < Kin; K0 += 32) {
      v16bf a, b;
#pragma unroll
      for (int i = 0; i < 16; i++) {
        a[i] = (bf16_t)af(K0 + a_k(i, g));
        b[i] = (n < nv) ? (bf16_t)W[(size_t)(K0 + b_k(i, g)) * Nout + (N0 + n)]
                        : (bf16_t)0.f;
      }
      c = wmma_bf16(a, b, c);
    }
#pragma unroll
    for (int r = 0; r < 8; r++) {
      int row = M0 + r + 8 * g;
      if (row < nrows && n < nv) {
        float x = c[r];
        if (act == 1) x = fmaxf(x, 0.f);
        else if (act == 2) x = (x > 0.f) ? x : 0.2f * x;
        out[(size_t)row * ld + (N0 + n)] = x;
      }
    }
  }
}

// ---------------- GNN kernels ----------------

// h = X @ Wi + bi   (N x 32 @ 32 x 64)
__global__ __launch_bounds__(256) void k_init(const float* __restrict__ X,
                                              const float* __restrict__ W,
                                              const float* __restrict__ b,
                                              float* __restrict__ h) {
  int wave = (blockIdx.x * blockDim.x + threadIdx.x) >> 5, lane = threadIdx.x & 31;
  int M0 = wave * 16;
  if (M0 >= NN) return;
  if (threadIdx.x == 0) prefetchW(W, FF * HH * 4);
  int ml = lane & 15, row = M0 + ml;
  if (row >= NN) row = NN - 1;
  const float* xr = X + (size_t)row * FF;
  layer_out([&](int k) { return xr[k]; },
            [&](int, int nn) { return b[nn]; },
            FF, W, HH, h, HH, M0, NN, 0, lane);
}

// msg = leaky(h[src] @ W, 0.2); agg[dst] += msg   (scatter via f32 atomics)
__global__ __launch_bounds__(256) void k_msg(const float* __restrict__ h,
                                             const int* __restrict__ adj,
                                             const float* __restrict__ W,
                                             float* __restrict__ agg) {
  int wave = (blockIdx.x * blockDim.x + threadIdx.x) >> 5, lane = threadIdx.x & 31;
  int M0 = wave * 16;
  if (M0 >= EE) return;
  if (threadIdx.x == 0) prefetchW(W, HH * HH * 4);
  int g = lane >> 4, ml = lane & 15, n = lane & 15;
  int e = M0 + ml;
  if (e >= EE) e = EE - 1;
  int src = adj[2 * e];
  const float* hr = h + (size_t)src * HH;
  v16bf a0, a1;
#pragma unroll
  for (int i = 0; i < 16; i++) {
    int k = a_k(i, g);
    a0[i] = (bf16_t)hr[k];
    a1[i] = (bf16_t)hr[32 + k];
  }
  int dst[8];
#pragma unroll
  for (int r = 0; r < 8; r++) {
    int row = M0 + r + 8 * g;
    dst[r] = (row < EE) ? adj[2 * row + 1] : -1;
  }
  v8f c[4];
#pragma unroll
  for (int t = 0; t < 4; t++)
#pragma unroll
    for (int r = 0; r < 8; r++) c[t][r] = 0.f;
#pragma unroll
  for (int t = 0; t < 4; t++) {
    v16bf b0, b1;
#pragma unroll
    for (int i = 0; i < 16; i++) {
      int k = b_k(i, g);
      b0[i] = (bf16_t)W[(size_t)k * HH + t * 16 + n];
      b1[i] = (bf16_t)W[(size_t)(32 + k) * HH + t * 16 + n];
    }
    c[t] = wmma_bf16(a0, b0, c[t]);
    c[t] = wmma_bf16(a1, b1, c[t]);
  }
#pragma unroll
  for (int t = 0; t < 4; t++)
#pragma unroll
    for (int r = 0; r < 8; r++) {
      if (dst[r] >= 0) {
        float x = c[t][r];
        x = (x > 0.f) ? x : 0.2f * x;
        atomicAdd(&agg[(size_t)dst[r] * HH + t * 16 + n], x);
      }
    }
}

// LayerNorm over 64 channels, one wave per node (wave32 shuffle reduction)
__global__ __launch_bounds__(256) void k_ln(const float* __restrict__ agg,
                                            float* __restrict__ h,
                                            const float* __restrict__ sc,
                                            const float* __restrict__ bi) {
  int wave = (blockIdx.x * blockDim.x + threadIdx.x) >> 5, lane = threadIdx.x & 31;
  if (wave >= NN) return;
  const float* r = agg + (size_t)wave * HH;
  float x0 = r[lane], x1 = r[lane + 32];
  float s = x0 + x1, s2 = x0 * x0 + x1 * x1;
#pragma unroll
  for (int o = 16; o > 0; o >>= 1) {
    s += __shfl_xor(s, o, 32);
    s2 += __shfl_xor(s2, o, 32);
  }
  float m = s * (1.f / 64.f);
  float var = s2 * (1.f / 64.f) - m * m;
  float inv = rsqrtf(var + 1e-5f);
  float* w = h + (size_t)wave * HH;
  w[lane] = sc[lane] * (x0 - m) * inv + bi[lane];
  w[lane + 32] = sc[lane + 32] * (x1 - m) * inv + bi[lane + 32];
}

// h = leaky(h @ Wd + bd) in place (A cached in regs before any store)
__global__ __launch_bounds__(256) void k_dense(float* __restrict__ h,
                                               const float* __restrict__ W,
                                               const float* __restrict__ b) {
  int wave = (blockIdx.x * blockDim.x + threadIdx.x) >> 5, lane = threadIdx.x & 31;
  int M0 = wave * 16;
  if (M0 >= NN) return;
  if (threadIdx.x == 0) prefetchW(W, HH * HH * 4);
  int g = lane >> 4, ml = lane & 15, n = lane & 15;
  int row = M0 + ml;
  if (row >= NN) row = NN - 1;
  const float* hr = h + (size_t)row * HH;
  v16bf a0, a1;
#pragma unroll
  for (int i = 0; i < 16; i++) {
    int k = a_k(i, g);
    a0[i] = (bf16_t)hr[k];
    a1[i] = (bf16_t)hr[32 + k];
  }
  v8f c[4];
#pragma unroll
  for (int t = 0; t < 4; t++)
#pragma unroll
    for (int r = 0; r < 8; r++) c[t][r] = b[t * 16 + n];
#pragma unroll
  for (int t = 0; t < 4; t++) {
    v16bf b0, b1;
#pragma unroll
    for (int i = 0; i < 16; i++) {
      int k = b_k(i, g);
      b0[i] = (bf16_t)W[(size_t)k * HH + t * 16 + n];
      b1[i] = (bf16_t)W[(size_t)(32 + k) * HH + t * 16 + n];
    }
    c[t] = wmma_bf16(a0, b0, c[t]);
    c[t] = wmma_bf16(a1, b1, c[t]);
  }
#pragma unroll
  for (int t = 0; t < 4; t++)
#pragma unroll
    for (int r = 0; r < 8; r++) {
      int rw = M0 + r + 8 * g;
      if (rw < NN) {
        float x = c[t][r];
        x = (x > 0.f) ? x : 0.2f * x;
        h[(size_t)rw * HH + t * 16 + n] = x;
      }
    }
}

// ---------------- Fused MLP heads ----------------

// node_repr(96) -> 128 -> 128 -> Nout (relu,relu,-); fused via wave-private LDS
__global__ __launch_bounds__(64) void k_node_mlp(
    const float* __restrict__ X, const float* __restrict__ h, const float* W0,
    const float* b0, const float* W1, const float* b1, const float* W2,
    const float* b2, int Nout, float* out) {
  __shared__ float sA[2][16 * 128];
  __shared__ float sB[2][16 * 128];
  int wv = threadIdx.x >> 5, lane = threadIdx.x & 31;
  int wave = blockIdx.x * 2 + wv, M0 = wave * 16;
  if (M0 >= NN) return;
  if (threadIdx.x == 0) { prefetchW(W0, 96 * 128 * 4); prefetchW(W1, 128 * 128 * 4); }
  float* bufA = sA[wv];
  float* bufB = sB[wv];
  int ml = lane & 15, row = M0 + ml;
  if (row >= NN) row = NN - 1;
  const float* xr = X + (size_t)row * FF;
  const float* hr = h + (size_t)row * HH;
  layer_fused<96, 128>([&](int k) { return (k < FF) ? xr[k] : hr[k - FF]; },
                       [&](int, int nn) { return b0[nn]; }, W0, bufA, 1, lane);
  layer_fused<128, 128>([&](int k) { return bufA[ml * 128 + k]; },
                        [&](int, int nn) { return b1[nn]; }, W1, bufB, 1, lane);
  layer_out([&](int k) { return bufB[ml * 128 + k]; },
            [&](int, int nn) { return b2[nn]; }, 128, W2, Nout, out, Nout, M0,
            NN, 0, lane);
}

// Per-graph precompute: pre[g] = [focus_NR | rep_avg | rep_sum | mrep] @ W + b0
__global__ __launch_bounds__(256) void k_pre(const float* __restrict__ X,
                                             const float* __restrict__ h,
                                             const int* __restrict__ focus,
                                             const float* __restrict__ ra,
                                             const float* __restrict__ rs,
                                             const float* __restrict__ mrep,
                                             const float* W, const float* b0,
                                             int withFocus, float* out) {
  int wave = (blockIdx.x * blockDim.x + threadIdx.x) >> 5, lane = threadIdx.x & 31;
  int M0 = wave * 16;
  if (M0 >= GG) return;
  int ml = lane & 15, grow = M0 + ml;
  if (grow >= GG) grow = GG - 1;
  const float* xr = X;
  const float* hr = h;
  if (withFocus) {
    int fn = focus[grow];
    xr = X + (size_t)fn * FF;
    hr = h + (size_t)fn * HH;
  }
  const float* rar = ra + (size_t)grow * HALFW;
  const float* rsr = rs + (size_t)grow * HALFW;
  const float* mr = mrep + (size_t)grow * MREPW;
  int Kin = withFocus ? 864 : 768;
  layer_out(
      [&](int k) {
        if (withFocus) {
          if (k < FF) return xr[k];
          if (k < 96) return hr[k - FF];
          k -= 96;
        }
        if (k < HALFW) return rar[k];
        if (k < 2 * HALFW) return rsr[k - HALFW];
        return mr[k - 2 * HALFW];
      },
      [&](int, int nn) { return b0[nn]; }, Kin, W, 128, out, 128, M0, GG, 0, lane);
}

// Edge head: t1 = NR[c1]@W0a + pre[g] + feat@W0f; relu; 128->64->32->Nout
__global__ __launch_bounds__(64) void k_edge_mlp(
    const float* __restrict__ X, const float* __restrict__ h,
    const int* __restrict__ ce, const float* __restrict__ feat,
    const int* __restrict__ n2g, const float* __restrict__ pre,
    const float* W0a, const float* W0f, const float* W1, const float* b1,
    const float* W2, const float* b2, const float* W3, const float* b3,
    int Nout, float* out) {
  __shared__ float sA[2][16 * 128];
  __shared__ float sB[2][16 * 64];
  int wv = threadIdx.x >> 5, lane = threadIdx.x & 31;
  int wave = blockIdx.x * 2 + wv, M0 = wave * 16;
  if (M0 >= CEN) return;
  if (threadIdx.x == 0) { prefetchW(W0a, 96 * 128 * 4); prefetchW(W1, 128 * 64 * 4); }
  float* bufA = sA[wv];
  float* bufB = sB[wv];
  int ml = lane & 15, e = M0 + ml;
  if (e >= CEN) e = CEN - 1;
  int c1 = ce[2 * e + 1];
  const float* xr = X + (size_t)c1 * FF;
  const float* hr = h + (size_t)c1 * HH;
  layer_fused<96, 128>(
      [&](int k) { return (k < FF) ? xr[k] : hr[k - FF]; },
      [&](int rr, int nn) {
        int er = M0 + rr;
        if (er >= CEN) er = CEN - 1;
        int gr = n2g[ce[2 * er]];
        const float* fe = feat + (size_t)er * 3;
        return pre[(size_t)gr * 128 + nn] + fe[0] * W0f[nn] +
               fe[1] * W0f[128 + nn] + fe[2] * W0f[256 + nn];
      },
      W0a, bufA, 1, lane);
  layer_fused<128, 64>([&](int k) { return bufA[ml * 128 + k]; },
                       [&](int, int nn) { return b1[nn]; }, W1, bufB, 1, lane);
  layer_fused<64, 32>([&](int k) { return bufB[ml * 64 + k]; },
                      [&](int, int nn) { return b2[nn]; }, W2, bufA, 1, lane);
  layer_out([&](int k) { return bufA[ml * 32 + k]; },
            [&](int, int nn) { return b3[nn]; }, 32, W3, Nout, out, Nout, M0,
            CEN, 0, lane);
}

// Attachment head: t1 = NR[ap]@Wa[0:96] + pre_at[g]; 128->64->32->1
__global__ __launch_bounds__(64) void k_attach(
    const float* __restrict__ X, const float* __restrict__ h,
    const int* __restrict__ ap, const int* __restrict__ n2g,
    const float* __restrict__ pre, const float* W0a, const float* W1,
    const float* b1, const float* W2, const float* b2, const float* W3,
    const float* b3, float* out) {
  __shared__ float sA[2][16 * 128];
  __shared__ float sB[2][16 * 64];
  int wv = threadIdx.x >> 5, lane = threadIdx.x & 31;
  int wave = blockIdx.x * 2 + wv, M0 = wave * 16;
  if (M0 >= APN) return;
  if (threadIdx.x == 0) { prefetchW(W0a, 96 * 128 * 4); prefetchW(W1, 128 * 64 * 4); }
  float* bufA = sA[wv];
  float* bufB = sB[wv];
  int ml = lane & 15, e = M0 + ml;
  if (e >= APN) e = APN - 1;
  int node = ap[e];
  const float* xr = X + (size_t)node * FF;
  const float* hr = h + (size_t)node * HH;
  layer_fused<96, 128>(
      [&](int k) { return (k < FF) ? xr[k] : hr[k - FF]; },
      [&](int rr, int nn) {
        int er = M0 + rr;
        if (er >= APN) er = APN - 1;
        int gr = n2g[ap[er]];
        return pre[(size_t)gr * 128 + nn];
      },
      W0a, bufA, 1, lane);
  layer_fused<128, 64>([&](int k) { return bufA[ml * 128 + k]; },
                       [&](int, int nn) { return b1[nn]; }, W1, bufB, 1, lane);
  layer_fused<64, 32>([&](int k) { return bufB[ml * 64 + k]; },
                      [&](int, int nn) { return b2[nn]; }, W2, bufA, 1, lane);
  layer_out([&](int k) { return bufA[ml * 32 + k]; },
            [&](int, int nn) { return b3[nn]; }, 32, W3, 1, out, 1, M0, APN, 0,
            lane);
}

// Stop head: t1 = relu(pre_sc[g] + nme_term); 128->64->32->1 (edge_scorer tail)
__global__ __launch_bounds__(64) void k_stop(const float* __restrict__ pre,
                                             const float* __restrict__ nt,
                                             const float* W1, const float* b1,
                                             const float* W2, const float* b2,
                                             const float* W3, const float* b3,
                                             float* out) {
  __shared__ float sA[2][16 * 128];
  __shared__ float sB[2][16 * 64];
  int wv = threadIdx.x >> 5, lane = threadIdx.x & 31;
  int wave = blockIdx.x * 2 + wv, M0 = wave * 16;
  if (M0 >= GG) return;
  float* bufA = sA[wv];
  float* bufB = sB[wv];
  int ml = lane & 15;
  for (int idx = lane; idx < 16 * 128; idx += 32) {
    int rr = idx >> 7, j = idx & 127;
    int gr = M0 + rr;
    if (gr >= GG) gr = GG - 1;
    float x = pre[(size_t)gr * 128 + j] + nt[j];
    bufA[idx] = fmaxf(x, 0.f);
  }
  layer_fused<128, 64>([&](int k) { return bufA[ml * 128 + k]; },
                       [&](int, int nn) { return b1[nn]; }, W1, bufB, 1, lane);
  layer_fused<64, 32>([&](int k) { return bufB[ml * 64 + k]; },
                      [&](int, int nn) { return b2[nn]; }, W2, bufA, 1, lane);
  layer_out([&](int k) { return bufA[ml * 32 + k]; },
            [&](int, int nn) { return b3[nn]; }, 32, W3, 1, out, 1, M0, GG, 0,
            lane);
}

// Node-selection head: graph_mol[grnc] (768) -> 256 -> 256 -> 65
__global__ __launch_bounds__(32) void k_nodesel(
    const int* __restrict__ grnc, const float* __restrict__ ra,
    const float* __restrict__ rs, const float* __restrict__ mrep,
    const float* W0, const float* b0, const float* W1, const float* b1,
    const float* W2, const float* b2, float* out) {
  __shared__ float sA[16 * 256];
  __shared__ float sB[16 * 256];
  int lane = threadIdx.x & 31;
  int M0 = blockIdx.x * 16;
  if (M0 >= NSEL) return;
  if (threadIdx.x == 0) { prefetchW(W0, 768 * 256 * 4); prefetchW(W1, 256 * 256 * 4); }
  int ml = lane & 15, row = M0 + ml;
  if (row >= NSEL) row = NSEL - 1;
  int g = grnc[row];
  const float* rar = ra + (size_t)g * HALFW;
  const float* rsr = rs + (size_t)g * HALFW;
  const float* mr = mrep + (size_t)g * MREPW;
  layer_fused<768, 256>(
      [&](int k) {
        if (k < HALFW) return rar[k];
        if (k < 2 * HALFW) return rsr[k - HALFW];
        return mr[k - 2 * HALFW];
      },
      [&](int, int nn) { return b0[nn]; }, W0, sA, 1, lane);
  layer_fused<256, 256>([&](int k) { return sA[ml * 256 + k]; },
                        [&](int, int nn) { return b1[nn]; }, W1, sB, 1, lane);
  layer_out([&](int k) { return sB[ml * 256 + k]; },
            [&](int, int nn) { return b2[nn]; }, 256, W2, NTL, out, NTL, M0,
            NSEL, 0, lane);
}

// ---------------- Segment / pooling kernels ----------------

__global__ void k_fill(float* p, float v, int n) {
  int i = blockIdx.x * blockDim.x + threadIdx.x;
  if (i < n) p[i] = v;
}
__global__ void k_fixfin(float* p, int n) {
  int i = blockIdx.x * blockDim.x + threadIdx.x;
  if (i < n) {
    float v = p[i];
    if (!isfinite(v)) p[i] = 0.f;
  }
}
__global__ void k_smax(const float* s, const int* n2g, float* gmax, int n) {
  int i = blockIdx.x * blockDim.x + threadIdx.x;
  if (i >= n) return;
  int node = i >> 3, hh = i & 7;
  atomicMaxF(&gmax[n2g[node] * HEADS + hh], s[i]);
}
__global__ void k_ssum(const float* s, const int* n2g, const float* gmax,
                       float* gsum, int n) {
  int i = blockIdx.x * blockDim.x + threadIdx.x;
  if (i >= n) return;
  int node = i >> 3, hh = i & 7;
  int g = n2g[node];
  atomicAdd(&gsum[g * HEADS + hh], expf(s[i] - gmax[g * HEADS + hh]));
}
__global__ void k_rep_avg(const float* s, const float* v, const int* n2g,
                          const float* gmax, const float* gsum, float* rep,
                          int total) {
  int i = blockIdx.x * blockDim.x + threadIdx.x;
  if (i >= total) return;
  int node = i >> 7, j = i & 127, hh = j >> 4;
  int g = n2g[node];
  float w = expf(s[node * HEADS + hh] - gmax[g * HEADS + hh]) /
            (gsum[g * HEADS + hh] + EPSF);
  atomicAdd(&rep[(size_t)g * HALFW + j], w * v[i]);
}
__global__ void k_rep_sum(const float* s, const float* v, const int* n2g,
                          float* rep, int total) {
  int i = blockIdx.x * blockDim.x + threadIdx.x;
  if (i >= total) return;
  int node = i >> 7, j = i & 127, hh = j >> 4;
  int g = n2g[node];
  float w = 1.f / (1.f + expf(-s[node * HEADS + hh]));
  float vv = UBF - softplusf(UBF - v[i]);
  atomicAdd(&rep[(size_t)g * HALFW + j], w * vv);
}
// nme_term[j] = nme[0:96]@W0[0:96,j] + nme[96:99]@W0[960:963,j]
__global__ void k_nme(const float* __restrict__ nme, const float* __restrict__ W0,
                      float* nt) {
  int j = threadIdx.x;
  float acc = 0.f;
  for (int k = 0; k < 96; k++) acc += nme[k] * W0[(size_t)k * 128 + j];
  for (int k = 0; k < 3; k++) acc += nme[96 + k] * W0[(size_t)(960 + k) * 128 + j];
  nt[j] = acc;
}
__global__ void k_amax(const float* att, const int* ap, const int* n2g,
                       float* amax, int n) {
  int i = blockIdx.x * blockDim.x + threadIdx.x;
  if (i >= n) return;
  atomicMaxF(&amax[n2g[ap[i]]], att[i]);
}
__global__ void k_asum(const float* att, const int* ap, const int* n2g,
                       const float* amax, float* asum, int n) {
  int i = blockIdx.x * blockDim.x + threadIdx.x;
  if (i >= n) return;
  int g = n2g[ap[i]];
  atomicAdd(&asum[g], expf(att[i] - amax[g]));
}
__global__ void k_aout(const float* att, const int* ap, const int* n2g,
                       const float* amax, const float* asum, float* out, int n) {
  int i = blockIdx.x * blockDim.x + threadIdx.x;
  if (i >= n) return;
  int g = n2g[ap[i]];
  out[i] = (att[i] - amax[g]) - logf(asum[g] + EPSF);
}

// ---------------------------------------------------------------------------
extern "C" void kernel_launch(void* const* d_in, const int* in_sizes, int n_in,
                              void* d_out, int out_size, void* d_ws,
                              size_t ws_size, hipStream_t stream) {
  (void)in_sizes; (void)n_in; (void)out_size; (void)ws_size;
  auto F32 = [&](int i) { return (const float*)d_in[i]; };
  auto I32 = [&](int i) { return (const int*)d_in[i]; };

  const float* X = F32(0);
  const int* adjs[3] = {I32(1), I32(2), I32(3)};
  const int* n2g = I32(4);
  const int* focus = I32(5);
  const int* apts = I32(6);
  const float* mrep = F32(7);
  const int* grnc = I32(8);
  const int* ce = I32(9);
  const float* cef = F32(10);

  const float *atW0 = F32(12), *atb0 = F32(13), *atW1 = F32(14), *atb1 = F32(15),
              *atW2 = F32(16), *atb2 = F32(17), *atW3 = F32(18), *atb3 = F32(19);
  const float *asW0 = F32(20), *asb0 = F32(21), *asW1 = F32(22), *asb1 = F32(23),
              *asW2 = F32(24), *asb2 = F32(25);
  const float *avW0 = F32(26), *avb0 = F32(27), *avW1 = F32(28), *avb1 = F32(29),
              *avW2 = F32(30), *avb2 = F32(31);
  const float *dW[3], *db[3];
  for (int i = 0; i < 3; i++) { dW[i] = F32(32 + 2 * i); db[i] = F32(33 + 2 * i); }
  const float *esW0 = F32(38), *esb0 = F32(39), *esW1 = F32(40), *esb1 = F32(41),
              *esW2 = F32(42), *esb2 = F32(43), *esW3 = F32(44), *esb3 = F32(45);
  const float *etW0 = F32(46), *etb0 = F32(47), *etW1 = F32(48), *etb1 = F32(49),
              *etW2 = F32(50), *etb2 = F32(51), *etW3 = F32(52), *etb3 = F32(53);
  const float *lnb[12], *lns[12];
  for (int i = 0; i < 12; i++) { lnb[i] = F32(54 + i); lns[i] = F32(66 + i); }
  const float* msgW[12][3];
  for (int l = 0; l < 12; l++)
    for (int t = 0; t < 3; t++) msgW[l][t] = F32(78 + l * 3 + t);
  const float* nme = F32(114);
  const float *nsW0 = F32(115), *nsb0 = F32(116), *nsW1 = F32(117),
              *nsb1 = F32(118), *nsW2 = F32(119), *nsb2 = F32(120);
  const float *ssW0 = F32(121), *ssb0 = F32(122), *ssW1 = F32(123),
              *ssb1 = F32(124), *ssW2 = F32(125), *ssb2 = F32(126);
  const float *stW0 = F32(127), *stb0 = F32(128), *stW1 = F32(129),
              *stb1 = F32(130), *stW2 = F32(131), *stb2 = F32(132);
  const float *wiW = F32(133), *wib = F32(134);

  // workspace carve (floats)
  float* ws = (float*)d_ws;
  size_t o = 0;
  float* h = ws + o;        o += (size_t)NN * HH;
  float* agg = ws + o;      o += (size_t)NN * HH;
  float* sbuf = ws + o;     o += (size_t)NN * HEADS;   // score buffer (reused)
  float* vbuf = ws + o;     o += (size_t)NN * HALFW;   // trans buffer (reused)
  float* gmax = ws + o;     o += (size_t)GG * HEADS;
  float* gsum = ws + o;     o += (size_t)GG * HEADS;
  float* rep_avg = ws + o;  o += (size_t)GG * HALFW;
  float* rep_sum = ws + o;  o += (size_t)GG * HALFW;
  float* pre_sc = ws + o;   o += (size_t)GG * HALFW;
  float* pre_ty = ws + o;   o += (size_t)GG * HALFW;
  float* pre_at = ws + o;   o += (size_t)GG * HALFW;
  float* nmet = ws + o;     o += 128;
  float* att = ws + o;      o += APN;
  float* amax = ws + o;     o += GG;
  float* asum = ws + o;     o += GG;

  const int nwN = (NN + 15) / 16, nwE = (EE + 15) / 16, nwG = (GG + 15) / 16;
  const int nwCE = (CEN + 15) / 16, nwAP = (APN + 15) / 16, nwNS = (NSEL + 15) / 16;

  // ---- GNN ----
  k_init<<<(nwN + 7) / 8, 256, 0, stream>>>(X, wiW, wib, h);
  for (int l = 0; l < LL; l++) {
    hipMemsetAsync(agg, 0, (size_t)NN * HH * sizeof(float), stream);
    for (int t = 0; t < TT; t++)
      k_msg<<<(nwE + 7) / 8, 256, 0, stream>>>(h, adjs[t], msgW[l][t], agg);
    k_ln<<<(NN + 7) / 8, 256, 0, stream>>>(agg, h, lns[l], lnb[l]);
    if ((l + 1) % 4 == 0)
      k_dense<<<(nwN + 7) / 8, 256, 0, stream>>>(h, dW[(l + 1) / 4 - 1],
                                                 db[(l + 1) / 4 - 1]);
  }

  // ---- avg pooling pipeline ----
  k_node_mlp<<<(nwN + 1) / 2, 64, 0, stream>>>(X, h, asW0, asb0, asW1, asb1,
                                               asW2, asb2, HEADS, sbuf);
  k_node_mlp<<<(nwN + 1) / 2, 64, 0, stream>>>(X, h, avW0, avb0, avW1, avb1,
                                               avW2, avb2, HALFW, vbuf);
  k_fill<<<(GG * HEADS + 255) / 256, 256, 0, stream>>>(gmax, -INFINITY, GG * HEADS);
  hipMemsetAsync(gsum, 0, (size_t)GG * HEADS * sizeof(float), stream);
  hipMemsetAsync(rep_avg, 0, (size_t)GG * HALFW * sizeof(float), stream);
  hipMemsetAsync(rep_sum, 0, (size_t)GG * HALFW * sizeof(float), stream);
  k_smax<<<(NN * HEADS + 255) / 256, 256, 0, stream>>>(sbuf, n2g, gmax, NN * HEADS);
  k_fixfin<<<(GG * HEADS + 255) / 256, 256, 0, stream>>>(gmax, GG * HEADS);
  k_ssum<<<(NN * HEADS + 255) / 256, 256, 0, stream>>>(sbuf, n2g, gmax, gsum,
                                                       NN * HEADS);
  k_rep_avg<<<(NN * HALFW + 255) / 256, 256, 0, stream>>>(sbuf, vbuf, n2g, gmax,
                                                          gsum, rep_avg, NN * HALFW);
  // ---- sum pooling pipeline (reuses sbuf/vbuf) ----
  k_node_mlp<<<(nwN + 1) / 2, 64, 0, stream>>>(X, h, ssW0, ssb0, ssW1, ssb1,
                                               ssW2, ssb2, HEADS, sbuf);
  k_node_mlp<<<(nwN + 1) / 2, 64, 0, stream>>>(X, h, stW0, stb0, stW1, stb1,
                                               stW2, stb2, HALFW, vbuf);
  k_rep_sum<<<(NN * HALFW + 255) / 256, 256, 0, stream>>>(sbuf, vbuf, n2g,
                                                          rep_sum, NN * HALFW);

  // ---- per-graph precomputes ----
  k_nme<<<1, 128, 0, stream>>>(nme, esW0, nmet);
  k_pre<<<(nwG + 7) / 8, 256, 0, stream>>>(X, h, focus, rep_avg, rep_sum, mrep,
                                           esW0 + 96 * 128, esb0, 1, pre_sc);
  k_pre<<<(nwG + 7) / 8, 256, 0, stream>>>(X, h, focus, rep_avg, rep_sum, mrep,
                                           etW0 + 96 * 128, etb0, 1, pre_ty);
  k_pre<<<(nwG + 7) / 8, 256, 0, stream>>>(X, h, focus, rep_avg, rep_sum, mrep,
                                           atW0 + 96 * 128, atb0, 0, pre_at);

  // ---- output heads ----
  float* out = (float*)d_out;
  float* out_ntl = out;                                  // 2500*65
  float* out_cand = out + (size_t)NSEL * NTL;            // CE
  float* out_stop = out_cand + CEN;                      // G
  float* out_etl = out_stop + GG;                        // CE*3
  float* out_att = out_etl + (size_t)CEN * 3;            // A

  k_nodesel<<<nwNS, 32, 0, stream>>>(grnc, rep_avg, rep_sum, mrep, nsW0, nsb0,
                                     nsW1, nsb1, nsW2, nsb2, out_ntl);
  k_edge_mlp<<<(nwCE + 1) / 2, 64, 0, stream>>>(
      X, h, ce, cef, n2g, pre_sc, esW0, esW0 + 960 * 128, esW1, esb1, esW2,
      esb2, esW3, esb3, 1, out_cand);
  k_stop<<<(nwG + 1) / 2, 64, 0, stream>>>(pre_sc, nmet, esW1, esb1, esW2, esb2,
                                           esW3, esb3, out_stop);
  k_edge_mlp<<<(nwCE + 1) / 2, 64, 0, stream>>>(
      X, h, ce, cef, n2g, pre_ty, etW0, etW0 + 960 * 128, etW1, etb1, etW2,
      etb2, etW3, etb3, 3, out_etl);
  k_attach<<<(nwAP + 1) / 2, 64, 0, stream>>>(X, h, apts, n2g, pre_at, atW0,
                                              atW1, atb1, atW2, atb2, atW3,
                                              atb3, att);
  k_fill<<<(GG + 255) / 256, 256, 0, stream>>>(amax, -INFINITY, GG);
  hipMemsetAsync(asum, 0, (size_t)GG * sizeof(float), stream);
  k_amax<<<(APN + 255) / 256, 256, 0, stream>>>(att, apts, n2g, amax, APN);
  k_fixfin<<<(GG + 255) / 256, 256, 0, stream>>>(amax, GG);
  k_asum<<<(APN + 255) / 256, 256, 0, stream>>>(att, apts, n2g, amax, asum, APN);
  k_aout<<<(APN + 255) / 256, 256, 0, stream>>>(att, apts, n2g, amax, asum,
                                                out_att, APN);
}
